// PodActor_22883585753322
// MI455X (gfx1250) — matrix operand order repaired and checked by reading, not compiled, not run on hardware
//
#include <hip/hip_runtime.h>
#include <hip/hip_bf16.h>
#include <math.h>

// Problem constants (match reference)
#define BB 2048
#define SS 256
#define HH 64
static constexpr int BS = BB * SS;

// d_out layout (flat concat of reference return tuple):
static constexpr size_t OFF_T   = 0;                      // thrust_mean [B,S,1]
static constexpr size_t OFF_A   = (size_t)BS;             // angle_mean  [B,S,1]
static constexpr size_t OFF_STD = (size_t)2 * BS;         // std         [B,S,2]
static constexpr size_t OFF_SH  = (size_t)4 * BS;         // shield      [B,S,2]
static constexpr size_t OFF_BO  = (size_t)6 * BS;         // boost       [B,S,2]
static constexpr size_t OFF_HN  = (size_t)8 * BS;         // hn [1,B,64]
static constexpr size_t OFF_CN  = (size_t)8 * BS + (size_t)BB * HH; // cn [1,B,64]

typedef __attribute__((ext_vector_type(16))) _Float16 v16h;
typedef __attribute__((ext_vector_type(8)))  _Float16 v8h;
typedef __attribute__((ext_vector_type(4)))  _Float16 v4h;
typedef __attribute__((ext_vector_type(8)))  float    v8f;

// Fast activations on hardware transcendentals (v_exp2 / v_rcp).
__device__ __forceinline__ float fexp(float x)  { return __builtin_amdgcn_exp2f(x * 1.4426950408889634f); }
__device__ __forceinline__ float frcp(float x)  { return __builtin_amdgcn_rcpf(x); }
__device__ __forceinline__ float fsigmoid(float x) { return frcp(1.f + fexp(-x)); }
__device__ __forceinline__ float ftanh(float x) { float e = fexp(2.f * x); return 1.f - 2.f * frcp(e + 1.f); }

__device__ __forceinline__ v8f wmma16(v16h a, v16h b, v8f c) {
  return __builtin_amdgcn_wmma_f32_16x16x32_f16(false, a, false, b, (short)0, c, false, false);
}

// A-matrix fragment (16x32 f16) from row-major LDS tile, leading dim lda (mult of 8).
// Contiguous k order used here (same as stage_b): lane&15 = row M; lanes 0-15 hold K 0..15, lanes 16-31 K 16..31.
// Two contiguous 16B runs per lane -> 2x ds_load_b128.
__device__ __forceinline__ v16h load_a_frag(const _Float16* As, int lda, int kt) {
  int lane = threadIdx.x & 31;
  const _Float16* base = As + (lane & 15) * lda + kt * 32 + ((lane >> 4) * 16);
  v8h lo = *(const v8h*)(base);
  v8h hi = *(const v8h*)(base + 8);
  return __builtin_shufflevector(lo, hi, 0, 1, 2, 3, 4, 5, 6, 7,
                                 8, 9, 10, 11, 12, 13, 14, 15);
}

// B-matrix fragment from PRE-SWIZZLED LDS: tile (kt,nt) stored as [32 lanes][16 halves]
// contiguous -> each lane reads 32 contiguous bytes (2x ds_load_b128).
__device__ __forceinline__ v16h load_b_frag_swz(const _Float16* Bs, int ntiles, int kt, int nt) {
  int lane = threadIdx.x & 31;
  const _Float16* p = Bs + (((kt * ntiles + nt) * 32 + lane) << 4);
  v8h lo = *(const v8h*)(p);
  v8h hi = *(const v8h*)(p + 8);
  return __builtin_shufflevector(lo, hi, 0, 1, 2, 3, 4, 5, 6, 7,
                                 8, 9, 10, 11, 12, 13, 14, 15);
}

// Stage a global f32 [K x N] row-major weight matrix into swizzled f16 fragments,
// zero-padding K up to Kpad (multiple of 32). N must be a multiple of 16.
template <int K, int N, int Kpad>
__device__ __forceinline__ void stage_b(_Float16* dst, const float* __restrict__ src,
                                        int tid, int nthreads) {
  constexpr int NT = N / 16;
  for (int i = tid; i < Kpad * N; i += nthreads) {
    int k = i / N, n = i % N;                  // N is power of two -> shifts
    float v = (k < K) ? src[k * N + n] : 0.f;
    int klo = k & 31;
    int lane = (n & 15) | (klo & 16);
    int e = klo & 15;
    dst[((((k >> 5) * NT + (n >> 4)) * 32 + lane) << 4) + e] = (_Float16)v;
  }
}

// C/D f32 16x16 layout: lane&15 = col N; VGPR r holds row M = r + 8*(lane>=16).
__device__ __forceinline__ void store_tile_f16(_Float16* D, int ldd, int dcol0, v8f c,
                                               const float* bias, int bcol0, bool relu) {
  int lane = threadIdx.x & 31;
  int col  = lane & 15;
  int rb   = (lane >> 4) * 8;
  float bv = bias[bcol0 + col];
#pragma unroll
  for (int r = 0; r < 8; ++r) {
    float v = c[r] + bv;
    if (relu) v = fmaxf(v, 0.f);
    D[(rb + r) * ldd + dcol0 + col] = (_Float16)v;
  }
}

// ---------------------------------------------------------------------------
// Kernel 1: token-parallel encoders + lstm_in assembly.
// Block = 128 threads = 4 waves; each wave owns a tile of 16 tokens.
// ---------------------------------------------------------------------------
__global__ __launch_bounds__(128) void encoder_kernel(
    const float* __restrict__ self_obs, const float* __restrict__ tm_obs,
    const float* __restrict__ en_obs,   const float* __restrict__ cp_obs,
    const int*   __restrict__ role_ids,
    const float* __restrict__ pe_w1, const float* __restrict__ pe_b1,
    const float* __restrict__ pe_w2, const float* __restrict__ pe_b2,
    const float* __restrict__ role_emb,
    const float* __restrict__ ee_w1, const float* __restrict__ ee_b1,
    const float* __restrict__ ee_w2, const float* __restrict__ ee_b2,
    const float* __restrict__ cb_w1, const float* __restrict__ cb_b1,
    const float* __restrict__ cb_w2, const float* __restrict__ cb_b2,
    _Float16* __restrict__ lstm_in) {
  // Swizzled weights (f16, K padded to multiples of 32), shared by all 4 waves.
  __shared__ __align__(16) _Float16 sPE1[32 * 64];
  __shared__ __align__(16) _Float16 sPE2[64 * 64];
  __shared__ __align__(16) _Float16 sEE1[32 * 32];
  __shared__ __align__(16) _Float16 sEE2[32 * 16];
  __shared__ __align__(16) _Float16 sCB1[64 * 128];
  __shared__ __align__(16) _Float16 sCB2[128 * 64];
  __shared__ float sbPE1[64], sbPE2[64], sbEE1[32], sbEE2[16], sbCB1[128], sbCB2[64];
  __shared__ float sRole[2 * 16];
  // Per-wave scratch (row-major A/M/OUT tiles)
  __shared__ __align__(16) _Float16 sA[4][16 * 64];
  __shared__ __align__(16) _Float16 sM[4][16 * 128];
  __shared__ __align__(16) _Float16 sOUT[4][16 * 128];
  __shared__ __align__(16) _Float16 sTM[4][16 * 16];
  __shared__ __align__(16) _Float16 sEN[4][16 * 16];

  const int tid  = threadIdx.x;
  const int wid  = tid >> 5;
  const int lane = tid & 31;

  stage_b<21, 64, 32>(sPE1, pe_w1, tid, 128);
  stage_b<64, 64, 64>(sPE2, pe_w2, tid, 128);
  stage_b<13, 32, 32>(sEE1, ee_w1, tid, 128);
  stage_b<32, 16, 32>(sEE2, ee_w2, tid, 128);
  stage_b<63, 128, 64>(sCB1, cb_w1, tid, 128);
  stage_b<128, 64, 128>(sCB2, cb_w2, tid, 128);
  if (tid < 64)  sbPE1[tid] = pe_b1[tid];
  if (tid < 64)  sbPE2[tid] = pe_b2[tid];
  if (tid < 32)  sbEE1[tid] = ee_b1[tid];
  if (tid < 16)  sbEE2[tid] = ee_b2[tid];
  if (tid < 128) sbCB1[tid] = cb_b1[tid];
  if (tid < 64)  sbCB2[tid] = cb_b2[tid];
  if (tid < 32)  sRole[tid] = role_emb[tid];
  __syncthreads();

  const int tile = blockIdx.x * 4 + wid;  // tile of 16 tokens
  const int tok0 = tile * 16;

  // ---- Pilot encoder: concat(self15, cp6) -> relu(21->64) -> relu(64->64) ----
  {
    int row = lane & 15, c0 = (lane >> 4) * 16, tok = tok0 + row;
#pragma unroll
    for (int j = 0; j < 16; ++j) {
      int col = c0 + j;
      float v = 0.f;
      if (col < 15)      v = self_obs[tok * 15 + col];
      else if (col < 21) v = cp_obs[tok * 6 + (col - 15)];
      sA[wid][row * 32 + col] = (_Float16)v;
    }
  }
  __syncthreads();
  {
    v16h a = load_a_frag(sA[wid], 32, 0);
#pragma unroll
    for (int nt = 0; nt < 4; ++nt) {
      v8f c = {};
      c = wmma16(a, load_b_frag_swz(sPE1, 4, 0, nt), c);
      store_tile_f16(sM[wid], 64, nt * 16, c, sbPE1, nt * 16, true);
    }
  }
  __syncthreads();
  {
    v16h afr[2];
#pragma unroll
    for (int kt = 0; kt < 2; ++kt) afr[kt] = load_a_frag(sM[wid], 64, kt);
#pragma unroll
    for (int nt = 0; nt < 4; ++nt) {   // p_emb -> sOUT cols 0..63
      v8f c = {};
#pragma unroll
      for (int kt = 0; kt < 2; ++kt) c = wmma16(afr[kt], load_b_frag_swz(sPE2, 4, kt, nt), c);
      store_tile_f16(sOUT[wid], 128, nt * 16, c, sbPE2, nt * 16, true);
    }
  }

  // ---- Entity encoder x3 (teammate, enemy0, enemy1); enemies max-pooled ----
  float enacc[8];
  for (int ent = 0; ent < 3; ++ent) {
    __syncthreads();
    {
      int row = lane & 15, c0 = (lane >> 4) * 16, tok = tok0 + row;
      const float* src = (ent == 0) ? (tm_obs + (size_t)tok * 13)
                                    : (en_obs + (size_t)(tok * 2 + (ent - 1)) * 13);
#pragma unroll
      for (int j = 0; j < 16; ++j) {
        int col = c0 + j;
        sA[wid][row * 32 + col] = (_Float16)(col < 13 ? src[col] : 0.f);
      }
    }
    __syncthreads();
    {
      v16h a = load_a_frag(sA[wid], 32, 0);
#pragma unroll
      for (int nt = 0; nt < 2; ++nt) {
        v8f c = {};
        c = wmma16(a, load_b_frag_swz(sEE1, 2, 0, nt), c);
        store_tile_f16(sM[wid], 32, nt * 16, c, sbEE1, nt * 16, true);
      }
    }
    __syncthreads();
    {
      v8f c = {};
      c = wmma16(load_a_frag(sM[wid], 32, 0), load_b_frag_swz(sEE2, 1, 0, 0), c);
      int col = lane & 15, rb = (lane >> 4) * 8;
      float bv = sbEE2[col];
      if (ent == 0) {
#pragma unroll
        for (int r = 0; r < 8; ++r) sTM[wid][(rb + r) * 16 + col] = (_Float16)(c[r] + bv);
      } else if (ent == 1) {
#pragma unroll
        for (int r = 0; r < 8; ++r) enacc[r] = c[r] + bv;
      } else {
#pragma unroll
        for (int r = 0; r < 8; ++r)
          sEN[wid][(rb + r) * 16 + col] = (_Float16)fmaxf(enacc[r], c[r] + bv);
      }
    }
  }
  __syncthreads();

  // ---- Command branch: concat(self15, tm16, en16, role16) -> relu(63->128) -> relu(128->64)
  {
    int row = lane & 15, c0 = (lane >> 4) * 32, tok = tok0 + row;
    int rid = role_ids[tok];
#pragma unroll
    for (int j = 0; j < 32; ++j) {
      int col = c0 + j;
      float v;
      if (col < 15)      v = self_obs[tok * 15 + col];
      else if (col < 31) v = (float)sTM[wid][row * 16 + (col - 15)];
      else if (col < 47) v = (float)sEN[wid][row * 16 + (col - 31)];
      else if (col < 63) v = sRole[rid * 16 + (col - 47)];
      else               v = 0.f;
      sA[wid][row * 64 + col] = (_Float16)v;
    }
  }
  __syncthreads();
  {
    v16h afr[2];
#pragma unroll
    for (int kt = 0; kt < 2; ++kt) afr[kt] = load_a_frag(sA[wid], 64, kt);
#pragma unroll
    for (int nt = 0; nt < 8; ++nt) {
      v8f c = {};
#pragma unroll
      for (int kt = 0; kt < 2; ++kt) c = wmma16(afr[kt], load_b_frag_swz(sCB1, 8, kt, nt), c);
      store_tile_f16(sM[wid], 128, nt * 16, c, sbCB1, nt * 16, true);
    }
  }
  __syncthreads();
  {
    v16h afr[4];
#pragma unroll
    for (int kt = 0; kt < 4; ++kt) afr[kt] = load_a_frag(sM[wid], 128, kt);
#pragma unroll
    for (int nt = 0; nt < 4; ++nt) {   // c_emb -> sOUT cols 64..127
      v8f c = {};
#pragma unroll
      for (int kt = 0; kt < 4; ++kt) c = wmma16(afr[kt], load_b_frag_swz(sCB2, 4, kt, nt), c);
      store_tile_f16(sOUT[wid], 128, 64 + nt * 16, c, sbCB2, nt * 16, true);
    }
  }
  __syncthreads();

  // ---- Write lstm_in (f16) [BS][128] ----
  {
    int row = lane & 15, tok = tok0 + row;
    const uint4* srcp = (const uint4*)(&sOUT[wid][row * 128]);
    uint4* dstp = (uint4*)(lstm_in + (size_t)tok * 128);
    int j0 = (lane >> 4) * 8;
#pragma unroll
    for (int j = 0; j < 8; ++j) dstp[j0 + j] = srcp[j0 + j];
  }
}

// ---------------------------------------------------------------------------
// Kernel 2: fused LSTM scan + output heads.
// Block = 256 threads = 8 waves; each block owns 16 batch rows for all S steps.
// Per step: gates = [lstm_in_t | h] (16x192) @ [ih_w; hh_w] (192x256) via WMMA;
// heads done with one extra WMMA pair on wave 0 reading h from A cols 128..191.
// ---------------------------------------------------------------------------
__global__ __launch_bounds__(256) void lstm_kernel(
    const _Float16* __restrict__ lstm_in,
    const float* __restrict__ ih_w, const float* __restrict__ ih_b,
    const float* __restrict__ hh_w, const float* __restrict__ hh_b,
    const float* __restrict__ th_w, const float* __restrict__ th_b,
    const float* __restrict__ an_w, const float* __restrict__ an_b,
    const float* __restrict__ sh_w, const float* __restrict__ sh_b,
    const float* __restrict__ bo_w, const float* __restrict__ bo_b,
    float* __restrict__ out) {
  __shared__ __align__(16) _Float16 sW[192 * 256];   // swizzled [ih_w ; hh_w], 96 KB
  __shared__ float sBias[256];                        // ih_b + hh_b
  __shared__ __align__(16) _Float16 sA[16 * 192];    // [lstm_in_t | h] per step
  __shared__ float sG[16 * 256];                      // gate pre-activations
  __shared__ __align__(16) _Float16 sHW[64 * 16];    // swizzled head weights (cols 6..15 = 0)
  __shared__ float sHB[16];

  const int tid = threadIdx.x;
  const int wid = tid >> 5;
  const int b0  = blockIdx.x * 16;

  // Stage gate weights swizzled: ih (K=128 -> ktiles 0..3), hh (K=64 -> ktiles 4,5).
  stage_b<128, 256, 128>(sW, ih_w, tid, 256);
  stage_b<64, 256, 64>(sW + 128 * 256, hh_w, tid, 256);
  if (tid < 256) sBias[tid] = ih_b[tid] + hh_b[tid];
  // Head weight matrix [64 x 16]: cols {th, an, sh0, sh1, bo0, bo1, 0...}, swizzled.
  for (int i = tid; i < 64 * 16; i += 256) {
    int k = i >> 4, n = i & 15;
    float v = 0.f;
    if (n == 0)      v = th_w[k];
    else if (n == 1) v = an_w[k];
    else if (n < 4)  v = sh_w[k * 2 + (n - 2)];
    else if (n < 6)  v = bo_w[k * 2 + (n - 4)];
    int klo = k & 31;
    sHW[((((k >> 5) * 1 + 0) * 32 + ((n & 15) | (klo & 16))) << 4) + (klo & 15)] = (_Float16)v;
  }
  if (tid < 16) {
    float v = 0.f;
    if (tid == 0) v = th_b[0];
    else if (tid == 1) v = an_b[0];
    else if (tid < 4) v = sh_b[tid - 2];
    else if (tid < 6) v = bo_b[tid - 4];
    sHB[tid] = v;
  }

  // Cell-state ownership: thread owns (row = tid/16, units u0..u0+3).
  const int row = tid >> 4;
  const int u0  = (tid & 15) * 4;
  float cst[4] = {0.f, 0.f, 0.f, 0.f};

  // Zero h part of A; load lstm_in for t=0 (8 halfs = 16 B per thread).
  for (int i = tid; i < 16 * 64; i += 256)
    sA[(i >> 6) * 192 + 128 + (i & 63)] = (_Float16)0.f;
  {
    int idx = tid * 8, r = idx >> 7, cc = idx & 127;
    *(uint4*)(&sA[r * 192 + cc]) =
        *(const uint4*)(lstm_in + ((size_t)(b0 + r) * SS + 0) * 128 + cc);
  }
  __syncthreads();

  for (int t = 0; t < SS; ++t) {
    // Phase 1: gate GEMM. A frags hoisted; wave wid owns N-tiles {2w, 2w+1}, K=192.
    {
      v16h afr[6];
#pragma unroll
      for (int kt = 0; kt < 6; ++kt) afr[kt] = load_a_frag(sA, 192, kt);
#pragma unroll
      for (int nn = 0; nn < 2; ++nn) {
        int nt = wid * 2 + nn;
        v8f c = {};
#pragma unroll
        for (int kt = 0; kt < 6; ++kt)
          c = wmma16(afr[kt], load_b_frag_swz(sW, 16, kt, nt), c);
        int lane = tid & 31, col = nt * 16 + (lane & 15), rb = (lane >> 4) * 8;
#pragma unroll
        for (int r = 0; r < 8; ++r) sG[(rb + r) * 256 + col] = c[r];
      }
    }
    __syncthreads();

    // Phase 2: elementwise LSTM update (vectorized LDS access, HW transcendentals).
    {
      const float* g = &sG[row * 256];
      float4 GI = *(const float4*)(g + u0);
      float4 GF = *(const float4*)(g + 64 + u0);
      float4 GG = *(const float4*)(g + 128 + u0);
      float4 GO = *(const float4*)(g + 192 + u0);
      float4 BI = *(const float4*)(&sBias[u0]);
      float4 BF = *(const float4*)(&sBias[64 + u0]);
      float4 BG = *(const float4*)(&sBias[128 + u0]);
      float4 BO = *(const float4*)(&sBias[192 + u0]);
      float gi[4] = {GI.x, GI.y, GI.z, GI.w};
      float gf[4] = {GF.x, GF.y, GF.z, GF.w};
      float gg[4] = {GG.x, GG.y, GG.z, GG.w};
      float go[4] = {GO.x, GO.y, GO.z, GO.w};
      float bi[4] = {BI.x, BI.y, BI.z, BI.w};
      float bf[4] = {BF.x, BF.y, BF.z, BF.w};
      float bg[4] = {BG.x, BG.y, BG.z, BG.w};
      float bo[4] = {BO.x, BO.y, BO.z, BO.w};
      v4h hq;
#pragma unroll
      for (int j = 0; j < 4; ++j) {
        float ig = fsigmoid(gi[j] + bi[j]);
        float fg = fsigmoid(gf[j] + bf[j]);
        float gv = ftanh(gg[j] + bg[j]);
        float og = fsigmoid(go[j] + bo[j]);
        float cc = fg * cst[j] + ig * gv;
        cst[j] = cc;
        hq[j] = (_Float16)(og * ftanh(cc));
      }
      *(v4h*)(&sA[row * 192 + 128 + u0]) = hq;   // h (f16) -> A k-tiles 4,5
    }
    __syncthreads();

    // Phase 3: heads via WMMA on wave 0 + prefetch next lstm_in tile.
    if (wid == 0) {
      v8f c = {};
      c = wmma16(load_a_frag(sA, 192, 4), load_b_frag_swz(sHW, 1, 0, 0), c);
      c = wmma16(load_a_frag(sA, 192, 5), load_b_frag_swz(sHW, 1, 1, 0), c);
      int lane = tid & 31, col = lane & 15, rb = (lane >> 4) * 8;
      if (col < 6) {
        float bv = sHB[col];
#pragma unroll
        for (int r = 0; r < 8; ++r) {
          float acc = c[r] + bv;
          size_t pos = (size_t)(b0 + rb + r) * SS + t;
          if (col == 0)      out[OFF_T + pos] = fsigmoid(acc);
          else if (col == 1) out[OFF_A + pos] = ftanh(acc);
          else if (col < 4)  out[OFF_SH + pos * 2 + (col - 2)] = acc;
          else               out[OFF_BO + pos * 2 + (col - 4)] = acc;
        }
      }
    }
    if (t + 1 < SS) {
      int idx = tid * 8, r = idx >> 7, cc = idx & 127;
      *(uint4*)(&sA[r * 192 + cc]) =
          *(const uint4*)(lstm_in + ((size_t)(b0 + r) * SS + (t + 1)) * 128 + cc);
    }
    __syncthreads();
  }

  // Final states hn/cn [1,B,64]; h read back from A h-section.
#pragma unroll
  for (int j = 0; j < 4; ++j) {
    int u = u0 + j;
    out[OFF_HN + (size_t)(b0 + row) * 64 + u] = (float)sA[row * 192 + 128 + u];
    out[OFF_CN + (size_t)(b0 + row) * 64 + u] = cst[j];
  }
}

// ---------------------------------------------------------------------------
// Kernel 3: constant std output.
// ---------------------------------------------------------------------------
__global__ __launch_bounds__(256) void std_fill_kernel(const float* __restrict__ logstd,
                                                       float* __restrict__ out) {
  int i = blockIdx.x * blockDim.x + threadIdx.x;
  if (i < 2 * BS) out[OFF_STD + i] = fmaxf(fexp(logstd[i & 1]), 0.05f);
}

extern "C" void kernel_launch(void* const* d_in, const int* in_sizes, int n_in,
                              void* d_out, int out_size, void* d_ws, size_t ws_size,
                              hipStream_t stream) {
  const float* self_obs = (const float*)d_in[0];
  const float* tm_obs   = (const float*)d_in[1];
  const float* en_obs   = (const float*)d_in[2];
  const float* cp_obs   = (const float*)d_in[3];
  const int*   role_ids = (const int*)d_in[4];
  const float* pe_w1 = (const float*)d_in[5],  *pe_b1 = (const float*)d_in[6];
  const float* pe_w2 = (const float*)d_in[7],  *pe_b2 = (const float*)d_in[8];
  const float* role_emb = (const float*)d_in[9];
  const float* ee_w1 = (const float*)d_in[10], *ee_b1 = (const float*)d_in[11];
  const float* ee_w2 = (const float*)d_in[12], *ee_b2 = (const float*)d_in[13];
  const float* cb_w1 = (const float*)d_in[14], *cb_b1 = (const float*)d_in[15];
  const float* cb_w2 = (const float*)d_in[16], *cb_b2 = (const float*)d_in[17];
  const float* ih_w  = (const float*)d_in[18], *ih_b  = (const float*)d_in[19];
  const float* hh_w  = (const float*)d_in[20], *hh_b  = (const float*)d_in[21];
  const float* th_w  = (const float*)d_in[22], *th_b  = (const float*)d_in[23];
  const float* an_w  = (const float*)d_in[24], *an_b  = (const float*)d_in[25];
  const float* sh_w  = (const float*)d_in[26], *sh_b  = (const float*)d_in[27];
  const float* bo_w  = (const float*)d_in[28], *bo_b  = (const float*)d_in[29];
  const float* logstd = (const float*)d_in[30];

  _Float16* ws_lstm = (_Float16*)d_ws;  // needs BS*128*2 = 128 MiB
  float* out = (float*)d_out;

  encoder_kernel<<<BS / 64, 128, 0, stream>>>(
      self_obs, tm_obs, en_obs, cp_obs, role_ids,
      pe_w1, pe_b1, pe_w2, pe_b2, role_emb,
      ee_w1, ee_b1, ee_w2, ee_b2,
      cb_w1, cb_b1, cb_w2, cb_b2, ws_lstm);

  lstm_kernel<<<BB / 16, 256, 0, stream>>>(
      ws_lstm, ih_w, ih_b, hh_w, hh_b,
      th_w, th_b, an_w, an_b, sh_w, sh_b, bo_w, bo_b, out);

  std_fill_kernel<<<(2 * BS + 255) / 256, 256, 0, stream>>>(logstd, out);
}
